// MegaNeRF_85899345920171
// MI455X (gfx1250) — compile-verified, matrix operand order checked by reading, not run on hardware
//
#include <hip/hip_runtime.h>

typedef __attribute__((ext_vector_type(16))) _Float16 v16h;
typedef __attribute__((ext_vector_type(8)))  float    v8f;

#define N_PTS  131072
#define NEXP   8
#define DIN    90
#define HID    256
#define NDOUT  4
#define MARGIN 2.0f
#define FEPS   1e-8f

#define KT 3            // ceil(90/32) K-tiles (padded to 96; k=90 = bias row)
#define NT (HID / 16)   // 16 N-tiles of the hidden dim
#define PACK_ELEMS (NEXP * KT * NT * 512)

// ---------------------------------------------------------------------------
// Pre-pass: pack W1 (E x DIN x H, f32) + b1 (E x H) into f16 WMMA-B-operand
// tiles in the exact v_wmma_f32_16x16x32_f16 B register layout (lane -> column
// N, K pairs across VGPRs; lanes 16-31 hold K+8 / K+24). The padded K row
// k==90 carries b1[e][h]; matching A rows carry 1.0, so the bias rides the
// matrix pipe and C starts as inline 0.
// ---------------------------------------------------------------------------
__global__ void pack_w1_kernel(const float* __restrict__ W1,
                               const float* __restrict__ b1,
                               _Float16* __restrict__ packed) {
  int idx = blockIdx.x * 256 + threadIdx.x;
  if (idx >= PACK_ELEMS) return;
  int j    = idx & 15;         // half index within lane
  int lane = (idx >> 4) & 31;  // lane
  int tile = idx >> 9;         // e*48 + kt*16 + nt
  int nt = tile & 15;
  int kt = (tile >> 4) % KT;
  int e  = tile / (KT * NT);
  int k = kt * 32 + ((lane & 16) ? 8 : 0) + (j < 8 ? j : j + 8);
  int h = nt * 16 + (lane & 15);
  float v;
  if (k < DIN)       v = W1[(e * DIN + k) * HID + h];
  else if (k == DIN) v = b1[e * HID + h];   // bias row
  else               v = 0.0f;
  packed[idx] = (_Float16)v;
}

// ---------------------------------------------------------------------------
// Fused MoE kernel: one wave32 processes 32 points (two 16-row A tiles), so
// every packed-W1 B tile feeds two WMMAs (halved B traffic per point).
//   gate weights (+ gate-weighted b2 term) -> LDS, one point per lane;
//   per expert: 16 N-tiles x 3 K-tiles x 2 WMMA (C = inline 0, bias via k=90),
//   relu(h*wgt) (wgt >= 0, so scale-then-clamp = clamp-then-scale and the
//   fmul output is canonical -> single v_max), layer-2 as packed FMAs,
//   single butterfly reduction at the end.
// ---------------------------------------------------------------------------
__global__ __launch_bounds__(256) void meganerf_kernel(
    const float* __restrict__ x,      // N x 93
    const float* __restrict__ cent,   // E x 3
    const _Float16* __restrict__ w1p, // packed W1 (+b1) tiles
    const float* __restrict__ W2,     // E x H x 4
    const float* __restrict__ b2,     // E x 4
    float* __restrict__ out)          // N x 4
{
  __shared__ float sW[8 * 32 * NEXP];  // [wave][point][expert] gate weights
  __shared__ float sB[8 * 32 * 4];     // [wave][point][o] sum_e w*b2
  const int lane = threadIdx.x & 31;
  const int wv   = threadIdx.x >> 5;
  const int base = blockIdx.x * 256 + wv * 32;  // first point of this wave
  const int l15  = lane & 15;
  const int hi   = lane >> 4;  // 0: rows 0-7, 1: rows 8-15 of C layout

  // ---- gate weights: one point per lane ----
  {
    const float* xp = x + (size_t)(base + lane) * 93;
    float px = xp[0], py = xp[1], pz = xp[2];
    float dist[NEXP];
    float mind = 3.4e38f;
#pragma unroll
    for (int e = 0; e < NEXP; ++e) {
      float dx = px - cent[e * 3 + 0];
      float dy = py - cent[e * 3 + 1];
      float dz = pz - cent[e * 3 + 2];
      float d2 = dx * dx + dy * dy + dz * dz;
      float d  = sqrtf(fmaxf(d2, 0.0f));
      dist[e] = d;
      mind = fminf(mind, d);
    }
    float s = 0.0f;
    float inv[NEXP];
#pragma unroll
    for (int e = 0; e < NEXP; ++e) {
      float iv = 1.0f / (dist[e] + FEPS);
      if (dist[e] > MARGIN * mind) iv = 0.0f;
      inv[e] = iv;
      s += iv;
    }
    float rs = 1.0f / s;
    float bt[4] = {0.f, 0.f, 0.f, 0.f};
#pragma unroll
    for (int e = 0; e < NEXP; ++e) {
      float w = inv[e] * rs;
      sW[(wv * 32 + lane) * NEXP + e] = w;
#pragma unroll
      for (int o = 0; o < 4; ++o) bt[o] = fmaf(w, b2[e * 4 + o], bt[o]);
    }
#pragma unroll
    for (int o = 0; o < 4; ++o) sB[(wv * 32 + lane) * 4 + o] = bt[o];
  }
  __syncthreads();

  // ---- A operands: two 16x96 feat tiles (f32 -> f16), loaded once.
  //      k == 90 is the constant-1.0 bias row.                        ----
  v16h a0[KT], a1[KT];
  {
    const float* fr0 = x + (size_t)(base + l15) * 93 + 3;
    const float* fr1 = x + (size_t)(base + 16 + l15) * 93 + 3;
#pragma unroll
    for (int kt = 0; kt < KT; ++kt) {
      int k0 = kt * 32 + (hi ? 8 : 0);
#pragma unroll
      for (int j = 0; j < 16; ++j) {
        int k = k0 + (j < 8 ? j : j + 8);
        _Float16 v0, v1;
        if (k < DIN)       { v0 = (_Float16)fr0[k]; v1 = (_Float16)fr1[k]; }
        else if (k == DIN) { v0 = (_Float16)1.0f;   v1 = (_Float16)1.0f;  }
        else               { v0 = (_Float16)0.0f;   v1 = (_Float16)0.0f;  }
        a0[kt][j] = v0;
        a1[kt][j] = v1;
      }
    }
  }

  float g0[8][4], g1[8][4];
#pragma unroll
  for (int r = 0; r < 8; ++r)
#pragma unroll
    for (int o = 0; o < 4; ++o) { g0[r][o] = 0.0f; g1[r][o] = 0.0f; }

  for (int e = 0; e < NEXP; ++e) {
    // gate weights for the 16 C-rows this lane contributes to, both tiles
    float wgt0[8], wgt1[8];
#pragma unroll
    for (int r = 0; r < 8; ++r) {
      wgt0[r] = sW[(wv * 32 + hi * 8 + r) * NEXP + e];
      wgt1[r] = sW[(wv * 32 + 16 + hi * 8 + r) * NEXP + e];
    }

    const _Float16* wbase = w1p + (size_t)e * KT * NT * 512;

#pragma unroll 4
    for (int nt = 0; nt < NT; ++nt) {
      v8f c0 = {}, c1 = {};
#pragma unroll
      for (int kt = 0; kt < KT; ++kt) {
        const v16h bv =
            *(const v16h*)(wbase + ((size_t)(kt * NT + nt)) * 512 + lane * 16);
        c0 = __builtin_amdgcn_wmma_f32_16x16x32_f16(
            false, a0[kt], false, bv, (short)0, c0, false, false);
        c1 = __builtin_amdgcn_wmma_f32_16x16x32_f16(
            false, a1[kt], false, bv, (short)0, c1, false, false);
      }

      // layer 2 with gate weight folded in: g += relu(h*wgt) * W2  (wgt >= 0)
      const float4 w2v =
          *(const float4*)(W2 + ((size_t)(e * HID + nt * 16 + l15)) * 4);
#pragma unroll
      for (int r = 0; r < 8; ++r) {
        float s0 = fmaxf(c0[r] * wgt0[r], 0.0f);
        float s1 = fmaxf(c1[r] * wgt1[r], 0.0f);
        g0[r][0] = fmaf(s0, w2v.x, g0[r][0]);
        g0[r][1] = fmaf(s0, w2v.y, g0[r][1]);
        g0[r][2] = fmaf(s0, w2v.z, g0[r][2]);
        g0[r][3] = fmaf(s0, w2v.w, g0[r][3]);
        g1[r][0] = fmaf(s1, w2v.x, g1[r][0]);
        g1[r][1] = fmaf(s1, w2v.y, g1[r][1]);
        g1[r][2] = fmaf(s1, w2v.z, g1[r][2]);
        g1[r][3] = fmaf(s1, w2v.w, g1[r][3]);
      }
    }
  }

  // ---- butterfly reduce across the 16-lane halves (wave32) ----
#pragma unroll
  for (int r = 0; r < 8; ++r)
#pragma unroll
    for (int o = 0; o < 4; ++o) {
      float v = g0[r][o];
      v += __shfl_xor(v, 1, 32);
      v += __shfl_xor(v, 2, 32);
      v += __shfl_xor(v, 4, 32);
      v += __shfl_xor(v, 8, 32);
      g0[r][o] = v;
      float u = g1[r][o];
      u += __shfl_xor(u, 1, 32);
      u += __shfl_xor(u, 2, 32);
      u += __shfl_xor(u, 4, 32);
      u += __shfl_xor(u, 8, 32);
      g1[r][o] = u;
    }

  // lanes 0 and 16 write 8 rows of each tile (+ precomputed b2 term)
  if (l15 == 0) {
#pragma unroll
    for (int r = 0; r < 8; ++r) {
      int m0 = hi * 8 + r;        // tile-0 local row
      int m1 = 16 + hi * 8 + r;   // tile-1 local row
      float4 o0, o1;
      o0.x = g0[r][0] + sB[(wv * 32 + m0) * 4 + 0];
      o0.y = g0[r][1] + sB[(wv * 32 + m0) * 4 + 1];
      o0.z = g0[r][2] + sB[(wv * 32 + m0) * 4 + 2];
      o0.w = g0[r][3] + sB[(wv * 32 + m0) * 4 + 3];
      o1.x = g1[r][0] + sB[(wv * 32 + m1) * 4 + 0];
      o1.y = g1[r][1] + sB[(wv * 32 + m1) * 4 + 1];
      o1.z = g1[r][2] + sB[(wv * 32 + m1) * 4 + 2];
      o1.w = g1[r][3] + sB[(wv * 32 + m1) * 4 + 3];
      *(float4*)(out + (size_t)(base + m0) * 4) = o0;
      *(float4*)(out + (size_t)(base + m1) * 4) = o1;
    }
  }
}

extern "C" void kernel_launch(void* const* d_in, const int* in_sizes, int n_in,
                              void* d_out, int out_size, void* d_ws,
                              size_t ws_size, hipStream_t stream) {
  const float* x    = (const float*)d_in[0];
  const float* cent = (const float*)d_in[1];
  const float* W1   = (const float*)d_in[2];
  const float* b1   = (const float*)d_in[3];
  const float* W2   = (const float*)d_in[4];
  const float* b2   = (const float*)d_in[5];
  float* out        = (float*)d_out;
  _Float16* packed  = (_Float16*)d_ws;  // 384 KB used

  pack_w1_kernel<<<(PACK_ELEMS + 255) / 256, 256, 0, stream>>>(W1, b1, packed);
  meganerf_kernel<<<N_PTS / 256, 256, 0, stream>>>(x, cent, packed, W2, b2,
                                                   out);
}
